// MultiHeadAttention_11321533792836
// MI455X (gfx1250) — matrix operand order repaired and checked by reading, not compiled, not run on hardware
//
#include <hip/hip_runtime.h>
#include <hip/hip_bf16.h>

typedef unsigned short u16;
typedef unsigned int   u32;
typedef __attribute__((ext_vector_type(16))) __bf16 v16bf;
typedef __attribute__((ext_vector_type(8)))  float  v8f;

union AB16 { v16bf v; u32 u[8]; u16 s[16]; };

#if defined(__has_builtin)
#if __has_builtin(__builtin_amdgcn_global_load_async_to_lds_b128)
#define USE_ASYNC_LDS 1
#endif
#endif
#ifndef USE_ASYNC_LDS
#define USE_ASYNC_LDS 0
#endif

// b128 async copy operand types: 16-byte int vector in global (AS1) / LDS (AS3)
typedef int v4i __attribute__((vector_size(16)));
typedef __attribute__((address_space(1))) v4i g_v4i;
typedef __attribute__((address_space(3))) v4i l_v4i;

__device__ __forceinline__ void wait_async_then_barrier() {
#if USE_ASYNC_LDS
#if __has_builtin(__builtin_amdgcn_s_wait_asynccnt)
    __builtin_amdgcn_s_wait_asynccnt(0);
#else
    asm volatile("s_wait_asynccnt 0" ::: "memory");
#endif
#endif
    __syncthreads();
}

__device__ __forceinline__ u16 f2bf(float f) {
    u32 u = __float_as_uint(f);
    u32 r = (u + 0x7FFFu + ((u >> 16) & 1u)) >> 16;   // round-to-nearest-even
    return (u16)r;
}

// ---------------------------------------------------------------------------
// fp32 -> bf16 conversion (one-time pass; everything downstream is bf16)
// ---------------------------------------------------------------------------
__global__ void cvt_f32_bf16(const float* __restrict__ in, u16* __restrict__ out, int n) {
    int i = blockIdx.x * 256 + threadIdx.x;
    if (i < n) out[i] = f2bf(in[i]);
}

// ---------------------------------------------------------------------------
// C[M,N] = A[M,1024] @ W[N,1024]^T + bias    (M=4096, N=1024)
// MODE 0: write bf16 into [B,H,T,64] head-major layout (QKV projections)
// MODE 1: write fp32 row-major to d_out (output projection)
// Block tile 128x128, 8 waves, each wave 32x64 via 2x4 WMMA 16x16x32 bf16.
// ---------------------------------------------------------------------------
template <int MODE>
__global__ __launch_bounds__(256) void gemm_bias(
    const u16* __restrict__ A, const u16* __restrict__ W,
    const float* __restrict__ bias, u16* __restrict__ outB, float* __restrict__ outF)
{
    constexpr int K = 1024, LDA = 1024;
    __shared__ u16 sA[128 * 40];   // [m][k] +8 pad
    __shared__ u16 sB[128 * 40];   // [n][k] +8 pad

    const int tid  = threadIdx.x;
    const int w    = tid >> 5, lane = tid & 31;
    const int hf   = lane >> 4, l16 = lane & 15;
    const int wm   = w & 3;        // 4 row groups of 32
    const int wn   = w >> 2;       // 2 col groups of 64
    const int rowBase = blockIdx.y * 128;
    const int colBase = blockIdx.x * 128;

    v8f acc[2][4];
    for (int i = 0; i < 2; ++i)
        for (int j = 0; j < 4; ++j)
            for (int e = 0; e < 8; ++e) acc[i][j][e] = 0.f;

    // staging coords: 4096 bf16 per tile = 512 x 16B chunks = 256 thr x 2
    const int sr0 = tid >> 2, sc0 = (tid & 3) * 8;
    const int sr1 = (tid + 256) >> 2, sc1 = ((tid + 256) & 3) * 8;

    for (int kk = 0; kk < K; kk += 32) {
        __syncthreads();            // protect LDS reuse
#if USE_ASYNC_LDS
        // CDNA5 async global->LDS (ASYNCcnt), no VGPR round-trip
        __builtin_amdgcn_global_load_async_to_lds_b128(
            (g_v4i*)(A + (size_t)(rowBase + sr0) * LDA + kk + sc0),
            (l_v4i*)(sA + sr0 * 40 + sc0), 0, 0);
        __builtin_amdgcn_global_load_async_to_lds_b128(
            (g_v4i*)(A + (size_t)(rowBase + sr1) * LDA + kk + sc1),
            (l_v4i*)(sA + sr1 * 40 + sc1), 0, 0);
        __builtin_amdgcn_global_load_async_to_lds_b128(
            (g_v4i*)(W + (size_t)(colBase + sr0) * LDA + kk + sc0),
            (l_v4i*)(sB + sr0 * 40 + sc0), 0, 0);
        __builtin_amdgcn_global_load_async_to_lds_b128(
            (g_v4i*)(W + (size_t)(colBase + sr1) * LDA + kk + sc1),
            (l_v4i*)(sB + sr1 * 40 + sc1), 0, 0);
#else
        *(uint4*)(sA + sr0 * 40 + sc0) = *(const uint4*)(A + (size_t)(rowBase + sr0) * LDA + kk + sc0);
        *(uint4*)(sA + sr1 * 40 + sc1) = *(const uint4*)(A + (size_t)(rowBase + sr1) * LDA + kk + sc1);
        *(uint4*)(sB + sr0 * 40 + sc0) = *(const uint4*)(W + (size_t)(colBase + sr0) * LDA + kk + sc0);
        *(uint4*)(sB + sr1 * 40 + sc1) = *(const uint4*)(W + (size_t)(colBase + sr1) * LDA + kk + sc1);
#endif
        wait_async_then_barrier();

        AB16 af[2], bw[4];
        #pragma unroll
        for (int i = 0; i < 2; ++i) {               // A-frag: 16x32, paired-K layout
            const u16* Ar = sA + (wm * 32 + i * 16 + l16) * 40;
            #pragma unroll
            for (int j = 0; j < 4; ++j) {
                int k0 = hf * 8 + j * 2;
                af[i].u[j]     = *(const u32*)(Ar + k0);
                af[i].u[j + 4] = *(const u32*)(Ar + 16 + k0);
            }
        }
        #pragma unroll
        for (int j = 0; j < 4; ++j) {               // B-frag: 32x16, paired-K layout
            const u16* Br = sB + (wn * 64 + j * 16 + l16) * 40 + hf * 8;
            ((uint4*)bw[j].s)[0] = *(const uint4*)(Br);
            ((uint4*)bw[j].s)[1] = *(const uint4*)(Br + 16);
        }
        #pragma unroll
        for (int i = 0; i < 2; ++i)
            #pragma unroll
            for (int j = 0; j < 4; ++j)
                acc[i][j] = __builtin_amdgcn_wmma_f32_16x16x32_bf16(
                    false, af[i].v, false, bw[j].v, (short)0, acc[i][j], false, false);
    }

    // epilogue: C layout row = r + 8*hf, col = l16
    #pragma unroll
    for (int i = 0; i < 2; ++i)
        #pragma unroll
        for (int j = 0; j < 4; ++j)
            #pragma unroll
            for (int r = 0; r < 8; ++r) {
                int mg = rowBase + wm * 32 + i * 16 + hf * 8 + r;
                int ng = colBase + wn * 64 + j * 16 + l16;
                float val = acc[i][j][r] + bias[ng];
                if (MODE == 0) {
                    int b = mg >> 11, t = mg & 2047;
                    int h = ng >> 6,  hd = ng & 63;
                    outB[(size_t)(((b * 16 + h) << 11) + t) * 64 + hd] = f2bf(val);
                } else {
                    outF[(size_t)mg * 1024 + ng] = val;
                }
            }
}

// ---------------------------------------------------------------------------
// Flash attention: Q,K,V in [B,H,2048,64] bf16 -> x in [B,2048,1024] bf16.
// Grid (TQ/128, B*H); 8 waves, each owns 16 query rows; stream 64-key tiles.
// ---------------------------------------------------------------------------
__global__ __launch_bounds__(256) void flash_attn(
    const u16* __restrict__ Qp, const u16* __restrict__ Kp,
    const u16* __restrict__ Vp, u16* __restrict__ xb)
{
    __shared__ u16 sK[64 * 72];        // [key][hd]
    __shared__ u16 sVt[64 * 72];       // [hd][key] (transposed for B-frags)
    __shared__ u16 sP[8 * 16 * 72];    // per-wave P staging (C-layout -> A-layout)

    const int tid = threadIdx.x;
    const int w = tid >> 5, lane = tid & 31;
    const int hf = lane >> 4, l16 = lane & 15;
    const int qt = blockIdx.x, bh = blockIdx.y;
    const int b = bh >> 4, h = bh & 15;
    const int q0 = qt * 128 + w * 16;

    const u16* Qh = Qp + (size_t)bh * 2048 * 64;
    const u16* Kh = Kp + (size_t)bh * 2048 * 64;
    const u16* Vh = Vp + (size_t)bh * 2048 * 64;

    // Q A-frags live in registers for the whole kernel (2 k-steps over HD=64)
    AB16 aq[2];
    {
        const u16* Qr = Qh + (size_t)(q0 + l16) * 64;
        #pragma unroll
        for (int s = 0; s < 2; ++s)
            #pragma unroll
            for (int j = 0; j < 4; ++j) {
                int k0 = s * 32 + hf * 8 + j * 2;
                aq[s].u[j]     = *(const u32*)(Qr + k0);
                aq[s].u[j + 4] = *(const u32*)(Qr + 16 + k0);
            }
    }

    float mrow[8], lrow[8];
    v8f O[4];
    #pragma unroll
    for (int r = 0; r < 8; ++r) { mrow[r] = -3.0e38f; lrow[r] = 0.f; }
    #pragma unroll
    for (int nt = 0; nt < 4; ++nt)
        for (int e = 0; e < 8; ++e) O[nt][e] = 0.f;

    const int kr0 = tid >> 3, kc0 = (tid & 7) * 8;               // K staging coords
    const int kr1 = (tid + 256) >> 3, kc1 = ((tid + 256) & 7) * 8;

    for (int kb = 0; kb < 32; ++kb) {
        const u16* Kt = Kh + (size_t)kb * 4096;
        const u16* Vt = Vh + (size_t)kb * 4096;
        __syncthreads();
        // stage K tile [64][64] via async global->LDS where available
#if USE_ASYNC_LDS
        __builtin_amdgcn_global_load_async_to_lds_b128(
            (g_v4i*)(Kt + kr0 * 64 + kc0), (l_v4i*)(sK + kr0 * 72 + kc0), 0, 0);
        __builtin_amdgcn_global_load_async_to_lds_b128(
            (g_v4i*)(Kt + kr1 * 64 + kc1), (l_v4i*)(sK + kr1 * 72 + kc1), 0, 0);
#else
        *(uint4*)(sK + kr0 * 72 + kc0) = *(const uint4*)(Kt + kr0 * 64 + kc0);
        *(uint4*)(sK + kr1 * 72 + kc1) = *(const uint4*)(Kt + kr1 * 64 + kc1);
#endif
        // stage V tile transposed -> [hd][key] (register path: transpose on store)
        for (int e = tid; e < 4096; e += 256) {
            int key = e >> 6, hd = e & 63;
            sVt[hd * 72 + key] = Vt[e];
        }
        wait_async_then_barrier();
        if (kb + 1 < 32) {                       // global_prefetch_b8 next tiles
            __builtin_prefetch(Kt + 4096, 0, 0);
            __builtin_prefetch(Vt + 4096, 0, 0);
        }

        // S = (Q @ K^T) * 1/sqrt(64)
        v8f S[4];
        #pragma unroll
        for (int nt = 0; nt < 4; ++nt) {
            for (int e = 0; e < 8; ++e) S[nt][e] = 0.f;
            #pragma unroll
            for (int s = 0; s < 2; ++s) {
                AB16 bk;
                const u16* Kr = sK + (nt * 16 + l16) * 72 + s * 32 + hf * 8;
                ((uint4*)bk.s)[0] = *(const uint4*)(Kr);
                ((uint4*)bk.s)[1] = *(const uint4*)(Kr + 16);
                S[nt] = __builtin_amdgcn_wmma_f32_16x16x32_bf16(
                    false, aq[s].v, false, bk.v, (short)0, S[nt], false, false);
            }
            for (int e = 0; e < 8; ++e) S[nt][e] *= 0.125f;
        }

        // online softmax per query row (row data lives across 16 lanes/half)
        u16* Pw = sP + w * 16 * 72;
        #pragma unroll
        for (int r = 0; r < 8; ++r) {
            float rm = fmaxf(fmaxf(S[0][r], S[1][r]), fmaxf(S[2][r], S[3][r]));
            #pragma unroll
            for (int off = 1; off <= 8; off <<= 1)
                rm = fmaxf(rm, __shfl_xor(rm, off, 32));
            float mn = fmaxf(mrow[r], rm);
            float al = __expf(mrow[r] - mn);
            mrow[r] = mn;
            float rs = 0.f;
            #pragma unroll
            for (int nt = 0; nt < 4; ++nt) {
                float p = __expf(S[nt][r] - mn);
                S[nt][r] = p;
                rs += p;
                Pw[(hf * 8 + r) * 72 + nt * 16 + l16] = f2bf(p);
            }
            #pragma unroll
            for (int off = 1; off <= 8; off <<= 1)
                rs += __shfl_xor(rs, off, 32);
            lrow[r] = lrow[r] * al + rs;
            #pragma unroll
            for (int nt = 0; nt < 4; ++nt) O[nt][r] *= al;
        }

        // reload P as A-frags (per-wave LDS region; same-wave dep only)
        AB16 ap[2];
        {
            const u16* Pr = Pw + l16 * 72;
            #pragma unroll
            for (int s = 0; s < 2; ++s)
                #pragma unroll
                for (int j = 0; j < 4; ++j) {
                    int k0 = s * 32 + hf * 8 + j * 2;
                    ap[s].u[j]     = *(const u32*)(Pr + k0);
                    ap[s].u[j + 4] = *(const u32*)(Pr + 16 + k0);
                }
        }

        // O += P @ V
        #pragma unroll
        for (int nt = 0; nt < 4; ++nt)
            #pragma unroll
            for (int s = 0; s < 2; ++s) {
                AB16 bv;
                const u16* Vr = sVt + (nt * 16 + l16) * 72 + s * 32 + hf * 8;
                ((uint4*)bv.s)[0] = *(const uint4*)(Vr);
                ((uint4*)bv.s)[1] = *(const uint4*)(Vr + 16);
                O[nt] = __builtin_amdgcn_wmma_f32_16x16x32_bf16(
                    false, ap[s].v, false, bv.v, (short)0, O[nt], false, false);
            }
    }

    // normalize and merge heads: x[b, t, h*64 + hd]
    #pragma unroll
    for (int r = 0; r < 8; ++r) {
        float inv = 1.0f / lrow[r];
        int t = q0 + hf * 8 + r;
        #pragma unroll
        for (int nt = 0; nt < 4; ++nt)
            xb[(size_t)(b * 2048 + t) * 1024 + h * 64 + nt * 16 + l16] = f2bf(O[nt][r] * inv);
    }
}

// ---------------------------------------------------------------------------
extern "C" void kernel_launch(void* const* d_in, const int* in_sizes, int n_in,
                              void* d_out, int out_size, void* d_ws, size_t ws_size,
                              hipStream_t stream)
{
    (void)in_sizes; (void)n_in; (void)out_size; (void)ws_size;
    const float* q  = (const float*)d_in[0];
    const float* k  = (const float*)d_in[1];
    const float* v  = (const float*)d_in[2];
    const float* Wq = (const float*)d_in[3];
    const float* bq = (const float*)d_in[4];
    const float* Wk = (const float*)d_in[5];
    const float* bk = (const float*)d_in[6];
    const float* Wv = (const float*)d_in[7];
    const float* bv = (const float*)d_in[8];
    const float* Wo = (const float*)d_in[9];
    const float* bo = (const float*)d_in[10];

    u16* ws = (u16*)d_ws;
    const size_t SZ = 4096u * 1024u;   // activation elements (B*T x 1024)
    const size_t WT = 1024u * 1024u;   // weight elements
    u16* qb  = ws;
    u16* kb_ = ws + SZ;
    u16* vb  = ws + 2 * SZ;
    u16* Wqb = ws + 3 * SZ;
    u16* Wkb = Wqb + WT;
    u16* Wvb = Wqb + 2 * WT;
    u16* Wob = Wqb + 3 * WT;
    u16* Qp  = Wqb + 4 * WT;
    u16* Kp  = Qp + SZ;
    u16* Vp  = Qp + 2 * SZ;
    u16* xb  = Qp + 3 * SZ;            // total 64 MB of workspace

    auto cv = [&](const float* src, u16* dst, size_t n) {
        cvt_f32_bf16<<<(int)((n + 255) / 256), 256, 0, stream>>>(src, dst, (int)n);
    };
    cv(q, qb, SZ);  cv(k, kb_, SZ);  cv(v, vb, SZ);
    cv(Wq, Wqb, WT); cv(Wk, Wkb, WT); cv(Wv, Wvb, WT); cv(Wo, Wob, WT);

    dim3 gg(8, 32), bb(256);
    gemm_bias<0><<<gg, bb, 0, stream>>>(qb,  Wqb, bq, Qp, nullptr);
    gemm_bias<0><<<gg, bb, 0, stream>>>(kb_, Wkb, bk, Kp, nullptr);
    gemm_bias<0><<<gg, bb, 0, stream>>>(vb,  Wvb, bv, Vp, nullptr);

    flash_attn<<<dim3(16, 32), bb, 0, stream>>>(Qp, Kp, Vp, xb);

    gemm_bias<1><<<gg, bb, 0, stream>>>(xb, Wob, bo, nullptr, (float*)d_out);
}
